// DeepSetEncoder_29420525978326
// MI455X (gfx1250) — matrix operand order repaired and checked by reading, not compiled, not run on hardware
//
#include <hip/hip_runtime.h>
#include <hip/hip_bf16.h>

// ---------------------------------------------------------------------------
// DeepSet encoder for MI455X (gfx1250, wave32, WMMA).
//   h  = relu(relu(x @ W1 + b1) @ W2 + b2)   [fused; x-tile + h1-tile live in LDS]
//   pooled = [sum_n h, max_n h]              [LDS reduce + global atomics]
//   out = relu(relu(pooled @ Wr1 + br1) @ Wr2 + br2)
// ---------------------------------------------------------------------------

typedef __attribute__((ext_vector_type(16))) __bf16 bf16x16;
typedef __attribute__((ext_vector_type(8)))  float  f32x8;
typedef __attribute__((ext_vector_type(4)))  float  f32x4;
typedef __attribute__((ext_vector_type(4)))  unsigned int u32x4;

#define DIM_B   64
#define DIM_N   512
#define DIM_IN  512
#define DIM_H   1024
#define ROWS_PER_BLOCK 64
#define XS_PAD  520             // 512 + 8  -> row stride 1040B (bank-conflict-free b128 A reads)
#define H1_PAD  1032            // 1024 + 8 -> row stride 2064B (bank-conflict-free b128 A reads)

union BF16Frag {
    bf16x16 v;
    __bf16  h[16];
    u32x4   q[2];
};

// ---- fragment loaders (layouts per CDNA5 ISA 7.12.2) ----------------------

// A 16x32 bf16 fragment from a bf16 row (LDS): lane (r=lane&15) holds row r;
// lanes<16: K = {k..k+7, k+16..k+23}; lanes>=16: K = {k+8..k+15, k+24..k+31}.
__device__ __forceinline__ bf16x16 load_a_bf16(const __bf16* rowp, int k, int hi) {
    const __bf16* p = rowp + k + hi * 8;
    BF16Frag f;
    f.q[0] = *(const u32x4*)(p);
    f.q[1] = *(const u32x4*)(p + 16);
    return f.v;
}

// B 32x16 bf16 fragment: lane holds column (col = base + lane&15); lanes<16:
// K={k..k+7, k+16..k+23}, lanes>=16: K={k+8..k+15, k+24..k+31}. Wt is [N][K] row-major.
__device__ __forceinline__ bf16x16 load_b_bf16(const __bf16* __restrict__ Wt,
                                               int col, int K, int k, int hi) {
    const __bf16* p = Wt + (size_t)col * K + k + hi * 8;
    BF16Frag f;
    f.q[0] = *(const u32x4*)(p);
    f.q[1] = *(const u32x4*)(p + 16);
    return f.v;
}

__device__ __forceinline__ f32x8 wmma_bf16(bf16x16 a, bf16x16 b, f32x8 c) {
    return __builtin_amdgcn_wmma_f32_16x16x32_bf16(false, a, false, b, (short)0, c, false, false);
}

// one pipeline stage: A fragments (2 row tiles) + B fragments (4 col tiles)
struct Frags {
    bf16x16 a[2];
    bf16x16 b[4];
};

template<int K>
__device__ __forceinline__ void load_frags(Frags& f,
                                           const __bf16* a0, const __bf16* a1,
                                           const __bf16* __restrict__ wt,
                                           int colBase, int r, int hi, int k) {
    f.a[0] = load_a_bf16(a0, k, hi);
    f.a[1] = load_a_bf16(a1, k, hi);
#pragma unroll
    for (int tn = 0; tn < 4; ++tn)
        f.b[tn] = load_b_bf16(wt, colBase + tn * 16 + r, K, k, hi);
}

__device__ __forceinline__ void mma_frags(const Frags& f, f32x8 acc[2][4]) {
#pragma unroll
    for (int tm = 0; tm < 2; ++tm)
#pragma unroll
        for (int tn = 0; tn < 4; ++tn)
            acc[tm][tn] = wmma_bf16(f.a[tm], f.b[tn], acc[tm][tn]);
}

// ---------------------------------------------------------------------------
// Ping-pong pipelined 32x64-per-wave GEMM tile: acc[2][4] += A[32xK] * B[Kx64].
// Two K-steps per iteration; buffers alternate roles structurally so no
// register rotation (no v_mov chains, no WMMA->VALU WAR nops).
// K/32 must be even (16 and 32 here).
// ---------------------------------------------------------------------------
template<int K>
__device__ __forceinline__ void gemm_32x64(const __bf16* a0, const __bf16* a1,
                                           const __bf16* __restrict__ wt,
                                           int colBase, int r, int hi,
                                           f32x8 acc[2][4]) {
    Frags f0, f1;
    load_frags<K>(f0, a0, a1, wt, colBase, r, hi, 0);            // step 0

#pragma unroll 1
    for (int k = 32; k + 32 < K; k += 64) {
        load_frags<K>(f1, a0, a1, wt, colBase, r, hi, k);        // step i+1 in flight
        mma_frags(f0, acc);                                      // step i
        load_frags<K>(f0, a0, a1, wt, colBase, r, hi, k + 32);   // step i+2 in flight
        mma_frags(f1, acc);                                      // step i+1
    }
    // tail: f0 holds step (K/32 - 2); last step K-32 goes through f1
    load_frags<K>(f1, a0, a1, wt, colBase, r, hi, K - 32);
    mma_frags(f0, acc);
    mma_frags(f1, acc);
}

// ---------------------------------------------------------------------------
// Weight transpose + fp32 -> bf16 convert:  W[K][N] fp32  ->  Wt[N][K] bf16
// ---------------------------------------------------------------------------
__global__ __launch_bounds__(256) void transpose_cvt(const float* __restrict__ W,
                                                     __bf16* __restrict__ Wt,
                                                     int K, int N) {
    __shared__ float tile[32][33];
    const int n0 = blockIdx.x * 32;
    const int k0 = blockIdx.y * 32;
    const int tx = threadIdx.x;      // 0..31
    const int ty = threadIdx.y;      // 0..7
#pragma unroll
    for (int j = 0; j < 32; j += 8)
        tile[ty + j][tx] = W[(size_t)(k0 + ty + j) * N + (n0 + tx)];
    __syncthreads();
#pragma unroll
    for (int j = 0; j < 32; j += 8)
        Wt[(size_t)(n0 + ty + j) * K + (k0 + tx)] = (__bf16)tile[tx][ty + j];
}

__global__ void init_pool(float* __restrict__ gsum, unsigned int* __restrict__ gmax, int n) {
    int i = blockIdx.x * blockDim.x + threadIdx.x;
    if (i < n) { gsum[i] = 0.0f; gmax[i] = 0u; }  // relu >= 0, so 0-bits is a valid max identity
}

// ---------------------------------------------------------------------------
// Fused phi MLP + pooling. One block = 64 set elements of one batch.
// 8 waves arranged 2 (row tiles of 32) x 4 (col groups of 64); column chunks
// of 256 iterate 4x to cover D_H = 1024.
// ---------------------------------------------------------------------------
__global__ __launch_bounds__(256) void phi_fused(const float* __restrict__ x,
                                                 const __bf16* __restrict__ w1t,
                                                 const float* __restrict__ b1,
                                                 const __bf16* __restrict__ w2t,
                                                 const float* __restrict__ b2,
                                                 float* __restrict__ gsum,
                                                 unsigned int* __restrict__ gmax) {
    __shared__ __align__(16) __bf16 xs[ROWS_PER_BLOCK][XS_PAD];    //  66.5 KB
    __shared__ __align__(16) __bf16 h1s[ROWS_PER_BLOCK][H1_PAD];   // 132.1 KB
    __shared__ float        sSum[DIM_H];                           //   4 KB
    __shared__ unsigned int sMax[DIM_H];                           //   4 KB

    const int tid  = threadIdx.x;
    const int lane = tid & 31;
    const int wid  = tid >> 5;      // 0..7
    const int wm   = wid & 1;       // row group: 32 rows each
    const int wn   = wid >> 1;      // col group: 64 cols each (of a 256 chunk)
    const int r    = lane & 15;
    const int hi   = lane >> 4;

    const int rowBase = blockIdx.x * ROWS_PER_BLOCK;   // global element row
    const int batch   = rowBase >> 9;                  // / DIM_N

    // ---- init pool LDS + stage x tile (fp32 -> bf16) into LDS -------------
    for (int i = tid; i < DIM_H; i += 256) { sSum[i] = 0.0f; sMax[i] = 0u; }
    for (int idx = tid; idx < (ROWS_PER_BLOCK * DIM_IN) / 4; idx += 256) {
        const int row = idx >> 7;                      // / (512/4)
        const int c4  = idx & 127;
        f32x4 v = *(const f32x4*)(x + (size_t)(rowBase + row) * DIM_IN + c4 * 4);
        union { __bf16 h[4]; unsigned long long u; } t;
        t.h[0] = (__bf16)v[0]; t.h[1] = (__bf16)v[1];
        t.h[2] = (__bf16)v[2]; t.h[3] = (__bf16)v[3];
        *(unsigned long long*)&xs[row][c4 * 4] = t.u;
    }
    __syncthreads();

    // ---------------- Phase 1: h1 = relu(x @ W1 + b1) -> LDS ---------------
    {
        const __bf16* a0 = &xs[wm * 32 + r][0];
        const __bf16* a1 = &xs[wm * 32 + 16 + r][0];
#pragma unroll 1
        for (int cc = 0; cc < 4; ++cc) {
            const int colBase = cc * 256 + wn * 64;
            f32x8 acc[2][4] = {};
            gemm_32x64<DIM_IN>(a0, a1, w1t, colBase, r, hi, acc);
#pragma unroll
            for (int tn = 0; tn < 4; ++tn) {
                const int col  = colBase + tn * 16 + r;
                const float bb = b1[col];
#pragma unroll
                for (int tm = 0; tm < 2; ++tm) {
#pragma unroll
                    for (int i = 0; i < 8; ++i) {
                        const int rr = wm * 32 + tm * 16 + i + hi * 8; // C: VGPR i -> M = i + 8*hi
                        float v = acc[tm][tn][i] + bb;
                        h1s[rr][col] = (__bf16)(v > 0.0f ? v : 0.0f);
                    }
                }
            }
        }
    }
    __syncthreads();

    // ------- Phase 2: h2 = relu(h1 @ W2 + b2), pooled (never stored) -------
    {
        const __bf16* a0 = &h1s[wm * 32 + r][0];
        const __bf16* a1 = &h1s[wm * 32 + 16 + r][0];
#pragma unroll 1
        for (int cc = 0; cc < 4; ++cc) {
            const int colBase = cc * 256 + wn * 64;
            f32x8 acc[2][4] = {};
            gemm_32x64<DIM_H>(a0, a1, w2t, colBase, r, hi, acc);
#pragma unroll
            for (int tn = 0; tn < 4; ++tn) {
                const int col  = colBase + tn * 16 + r;
                const float bb = b2[col];
                float s = 0.0f, m = 0.0f;
#pragma unroll
                for (int tm = 0; tm < 2; ++tm) {
#pragma unroll
                    for (int i = 0; i < 8; ++i) {
                        float v = acc[tm][tn][i] + bb;
                        v = v > 0.0f ? v : 0.0f;
                        s += v;
                        m = fmaxf(m, v);
                    }
                }
                // combine row-halves (lane L <-> L+16): 32 rows total per wave
                s += __shfl_xor(s, 16, 32);
                m  = fmaxf(m, __shfl_xor(m, 16, 32));
                if (lane < 16) {
                    atomicAdd(&sSum[col], s);
                    atomicMax(&sMax[col], __float_as_uint(m));  // valid: values >= 0
                }
            }
        }
    }
    __syncthreads();

    // ---------------- block partials -> global pool buffers ----------------
    for (int i = tid; i < DIM_H; i += 256) {
        atomicAdd(&gsum[batch * DIM_H + i], sSum[i]);
        atomicMax(&gmax[batch * DIM_H + i], sMax[i]);
    }
}

// ---------------------------------------------------------------------------
// rho layers: tiny fp32 GEMMs (64 rows). Each thread: one column, 8 rows.
// ---------------------------------------------------------------------------
__global__ __launch_bounds__(256) void rho1_kernel(const float* __restrict__ gsum,
                                                   const unsigned int* __restrict__ gmax,
                                                   const float* __restrict__ W,   // [2048][1024]
                                                   const float* __restrict__ bias,
                                                   float* __restrict__ out) {     // [64][1024]
    const int n  = blockIdx.x * 256 + threadIdx.x;  // grid.x = 4
    const int m0 = blockIdx.y * 8;                  // grid.y = 8
    float acc[8] = {};
    for (int k = 0; k < DIM_H; ++k) {
        const float w = W[(size_t)k * DIM_H + n];
#pragma unroll
        for (int mm = 0; mm < 8; ++mm)
            acc[mm] = fmaf(gsum[(m0 + mm) * DIM_H + k], w, acc[mm]);
    }
    for (int k = 0; k < DIM_H; ++k) {
        const float w = W[(size_t)(DIM_H + k) * DIM_H + n];
#pragma unroll
        for (int mm = 0; mm < 8; ++mm)
            acc[mm] = fmaf(__uint_as_float(gmax[(m0 + mm) * DIM_H + k]), w, acc[mm]);
    }
    const float b = bias[n];
#pragma unroll
    for (int mm = 0; mm < 8; ++mm) {
        float v = acc[mm] + b;
        out[(m0 + mm) * DIM_H + n] = v > 0.0f ? v : 0.0f;
    }
}

__global__ __launch_bounds__(256) void rho2_kernel(const float* __restrict__ in,  // [64][1024]
                                                   const float* __restrict__ W,   // [1024][1024]
                                                   const float* __restrict__ bias,
                                                   float* __restrict__ out) {     // [64][1024]
    const int n  = blockIdx.x * 256 + threadIdx.x;
    const int m0 = blockIdx.y * 8;
    float acc[8] = {};
    for (int k = 0; k < DIM_H; ++k) {
        const float w = W[(size_t)k * DIM_H + n];
#pragma unroll
        for (int mm = 0; mm < 8; ++mm)
            acc[mm] = fmaf(in[(m0 + mm) * DIM_H + k], w, acc[mm]);
    }
    const float b = bias[n];
#pragma unroll
    for (int mm = 0; mm < 8; ++mm) {
        float v = acc[mm] + b;
        out[(m0 + mm) * DIM_H + n] = v > 0.0f ? v : 0.0f;
    }
}

// ---------------------------------------------------------------------------
extern "C" void kernel_launch(void* const* d_in, const int* in_sizes, int n_in,
                              void* d_out, int out_size, void* d_ws, size_t ws_size,
                              hipStream_t stream) {
    (void)in_sizes; (void)n_in; (void)out_size; (void)ws_size;

    const float* x   = (const float*)d_in[0];
    const float* W1  = (const float*)d_in[1];
    const float* b1  = (const float*)d_in[2];
    const float* W2  = (const float*)d_in[3];
    const float* b2  = (const float*)d_in[4];
    const float* Wr1 = (const float*)d_in[5];
    const float* br1 = (const float*)d_in[6];
    const float* Wr2 = (const float*)d_in[7];
    const float* br2 = (const float*)d_in[8];
    float* out = (float*)d_out;

    // workspace carve-up (~3.9 MB total)
    char* ws = (char*)d_ws;
    __bf16*       w1t  = (__bf16*)(ws);                                  // 1024*512*2  = 1,048,576
    __bf16*       w2t  = (__bf16*)(ws + 1048576);                        // 1024*1024*2 = 2,097,152
    float*        gsum = (float*)(ws + 1048576 + 2097152);               // 64*1024*4   =   262,144
    unsigned int* gmax = (unsigned int*)(ws + 1048576 + 2097152 + 262144);
    float*        r1   = (float*)(ws + 1048576 + 2097152 + 2 * 262144);

    // 1) weights -> bf16, transposed to [N][K]
    transpose_cvt<<<dim3(DIM_H / 32, DIM_IN / 32), dim3(32, 8), 0, stream>>>(W1, w1t, DIM_IN, DIM_H);
    transpose_cvt<<<dim3(DIM_H / 32, DIM_H / 32),  dim3(32, 8), 0, stream>>>(W2, w2t, DIM_H, DIM_H);

    // 2) zero pooling accumulators
    init_pool<<<(DIM_B * DIM_H + 255) / 256, 256, 0, stream>>>(gsum, gmax, DIM_B * DIM_H);

    // 3) fused phi MLP + sum/max pooling
    phi_fused<<<(DIM_B * DIM_N) / ROWS_PER_BLOCK, 256, 0, stream>>>(x, w1t, b1, w2t, b2, gsum, gmax);

    // 4) rho MLP
    rho1_kernel<<<dim3(4, 8), 256, 0, stream>>>(gsum, gmax, Wr1, br1, r1);
    rho2_kernel<<<dim3(4, 8), 256, 0, stream>>>(r1, Wr2, br2, out);
}
